// Transformer_15470472200589
// MI455X (gfx1250) — matrix operand order repaired and checked by reading, not compiled, not run on hardware
//
#include <hip/hip_runtime.h>
#include <hip/hip_bf16.h>

// ---------------- model dims (from reference) ----------------
constexpr int LNUM = 4;
constexpr int EMB  = 1024;
constexpr int HID  = 1024;
constexpr int NH   = 16;
constexpr int VOC  = 32000;
constexpr int SEQ  = 1024;
constexpr int BAT  = 2;
constexpr int HD   = HID / NH;     // 64
constexpr int MTOT = BAT * SEQ;    // 2048
constexpr int FFD  = 4 * EMB;      // 4096

typedef __attribute__((ext_vector_type(16))) __bf16        v16bf;
typedef __attribute__((ext_vector_type(8)))  unsigned int  v8u;
typedef __attribute__((ext_vector_type(8)))  float         v8f;

union U16x8 { uint4 v; unsigned short u[8]; unsigned int w[4]; };

__device__ __forceinline__ unsigned short f2bf(float f) {
  unsigned int u = __builtin_bit_cast(unsigned int, f);
  u += 0x7FFFu + ((u >> 16) & 1u);            // round-to-nearest-even
  return (unsigned short)(u >> 16);
}

// ---------------- generic batched/strided bf16 GEMM with WMMA ----------------
// C[z][m,n] = alpha * sum_k A[z][m,k]*B[z][k,n] (+bias[n]) (relu?)
// A/B bf16 (raw u16), C fp32. z -> (z/nInner, z%nInner) batch bases.
// Block = 256 threads (8 waves), tile 128x128x64, wave tile 64x32.
// 16 v_wmma_f32_16x16x32_bf16 per stage/barrier pair (two K=32 sub-steps).
// Requirements used by all call sites: M%128==0, K%64==0, N%8==0; one of
// {csA==1, rsA==1} and one of {csB==1, rsB==1}, all strides/bases %8==0
// (16B-aligned vector loads). Generic scalar path kept as fallback.
__global__ __launch_bounds__(256) void gemm_bf16_wmma(
    const unsigned short* __restrict__ A,
    const unsigned short* __restrict__ Bm,
    float* __restrict__ C,
    const float* __restrict__ bias,
    int M, int N, int K,
    long baoA, long baiA, long rsA, long csA,
    long baoB, long baiB, long rsB, long csB,
    long baoC, long baiC, long ldc,
    int nInner, float alpha, int relu)
{
  // A tile as dwords: row m holds K-pairs (k even,k+1) -> dword k/2 (0..31); pad 33
  __shared__ unsigned int As32[128][33];
  // B tile pre-interleaved pair-rows: Bs32[k/2][n] = {B[k+1][n], B[k][n]}
  __shared__ unsigned int Bs32[32][132];

  const int tid  = threadIdx.x;
  const int wid  = tid >> 5;
  const int lane = tid & 31;
  const int waveRow = wid >> 2;            // 0..1  -> 64-row slab
  const int waveCol = wid & 3;             // 0..3  -> 32-col slab
  const int h  = lane >> 4;                // lane half
  const int ml = lane & 15;

  const int tileM = blockIdx.y * 128;
  const int tileN = blockIdx.x * 128;
  const long zo = blockIdx.z / nInner;
  const long zi = blockIdx.z % nInner;
  const long baseA = zo * baoA + zi * baiA;
  const long baseB = zo * baoB + zi * baiB;
  const long baseC = zo * baoC + zi * baiC;

  unsigned short* AsU = (unsigned short*)&As32[0][0];  // row stride 66 u16
  unsigned short* BsU = (unsigned short*)&Bs32[0][0];  // (kp*132+n)*2 + (k&1)

  v8f acc[4][2] = {};

  for (int k0 = 0; k0 < K; k0 += 64) {
    // ================= stage A tile (128 x 64 bf16) =================
    if (csA == 1) {
      // vector along K: 128 rows x 8 chunks(8 elem) = 1024 -> 4/thread
      for (int i = tid; i < 1024; i += 256) {
        int m = i >> 3, c = i & 7;
        U16x8 d;
        d.v = *(const uint4*)(A + baseA + (long)(tileM + m) * rsA + (k0 + c * 8));
        As32[m][c * 4 + 0] = d.w[0];
        As32[m][c * 4 + 1] = d.w[1];
        As32[m][c * 4 + 2] = d.w[2];
        As32[m][c * 4 + 3] = d.w[3];
        if (k0 + 64 < K)   // prefetch next K-tile (-> global_prefetch_b8)
          __builtin_prefetch(A + baseA + (long)(tileM + m) * rsA + (k0 + 64 + c * 8), 0, 3);
      }
    } else if (rsA == 1) {
      // vector along M (transposed A, e.g. attn^T): 64 k x 16 chunks(8 m)
      for (int i = tid; i < 1024; i += 256) {
        int kk = i >> 4, m0 = (i & 15) * 8;
        U16x8 d;
        d.v = *(const uint4*)(A + baseA + (long)(k0 + kk) * csA + (tileM + m0));
        #pragma unroll
        for (int q = 0; q < 8; q++)
          AsU[(m0 + q) * 66 + kk] = d.u[q];
        if (k0 + 64 < K)
          __builtin_prefetch(A + baseA + (long)(k0 + 64 + kk) * csA + (tileM + m0), 0, 3);
      }
    } else {
      for (int i = tid; i < 128 * 64; i += 256) {
        int m = i >> 6, kk = i & 63;
        int gm = tileM + m, gk = k0 + kk;
        unsigned short v = 0;
        if (gm < M && gk < K) v = A[baseA + (long)gm * rsA + (long)gk * csA];
        AsU[m * 66 + kk] = v;
      }
    }

    // ================= stage B tile (64 x 128 bf16) =================
    if (csB == 1) {
      // vector along N: 32 pair-rows x 16 chunks(8 n) = 512 -> 2/thread
      for (int i = tid; i < 512; i += 256) {
        int kp = i >> 4, n0 = (i & 15) * 8;
        int gn = tileN + n0, gk = k0 + kp * 2;
        U16x8 lo, hi;
        lo.v = make_uint4(0, 0, 0, 0);
        hi.v = make_uint4(0, 0, 0, 0);
        if (gn < N) {
          lo.v = *(const uint4*)(Bm + baseB + (long)gk * rsB + gn);
          hi.v = *(const uint4*)(Bm + baseB + (long)(gk + 1) * rsB + gn);
          if (k0 + 64 < K)
            __builtin_prefetch(Bm + baseB + (long)(gk + 64) * rsB + gn, 0, 3);
        }
        uint4 o0, o1;
        o0.x = lo.u[0] | ((unsigned)hi.u[0] << 16);
        o0.y = lo.u[1] | ((unsigned)hi.u[1] << 16);
        o0.z = lo.u[2] | ((unsigned)hi.u[2] << 16);
        o0.w = lo.u[3] | ((unsigned)hi.u[3] << 16);
        o1.x = lo.u[4] | ((unsigned)hi.u[4] << 16);
        o1.y = lo.u[5] | ((unsigned)hi.u[5] << 16);
        o1.z = lo.u[6] | ((unsigned)hi.u[6] << 16);
        o1.w = lo.u[7] | ((unsigned)hi.u[7] << 16);
        *(uint4*)&Bs32[kp][n0]     = o0;   // rows are 528B -> 16B aligned
        *(uint4*)&Bs32[kp][n0 + 4] = o1;
      }
    } else if (rsB == 1) {
      // vector along K: 128 n x 8 chunks(8 k) = 1024 -> 4/thread
      for (int i = tid; i < 1024; i += 256) {
        int n = i >> 3, kk0 = (i & 7) * 8;
        int gn = tileN + n;
        U16x8 d;
        d.v = make_uint4(0, 0, 0, 0);
        if (gn < N) {
          d.v = *(const uint4*)(Bm + baseB + (long)gn * csB + (k0 + kk0));
          if (k0 + 64 < K)
            __builtin_prefetch(Bm + baseB + (long)gn * csB + (k0 + 64 + kk0), 0, 3);
        }
        #pragma unroll
        for (int q = 0; q < 4; q++)
          Bs32[(kk0 >> 1) + q][n] = d.u[2 * q] | ((unsigned)d.u[2 * q + 1] << 16);
      }
    } else {
      for (int i = tid; i < 64 * 128; i += 256) {
        int kk = i >> 7, n = i & 127;
        int gn = tileN + n, gk = k0 + kk;
        unsigned short v = 0;
        if (gn < N && gk < K) v = Bm[baseB + (long)gk * rsB + (long)gn * csB];
        BsU[((kk >> 1) * 132 + n) * 2 + (kk & 1)] = v;
      }
    }
    __syncthreads();

    // ---- two K=32 sub-steps: 16 WMMAs per barrier pair ----
    #pragma unroll
    for (int s = 0; s < 2; s++) {
      v16bf fa[4];
      #pragma unroll
      for (int i = 0; i < 4; i++) {
        int mo = waveRow * 64 + i * 16 + ml;
        v8u t;
        #pragma unroll
        for (int v = 0; v < 8; v++) {
          int idx = s * 16 + 8 * h + v;
          t[v] = As32[mo][idx];
        }
        fa[i] = __builtin_bit_cast(v16bf, t);
      }
      v16bf fb[2];
      #pragma unroll
      for (int j = 0; j < 2; j++) {
        int no = waveCol * 32 + j * 16 + ml;
        v8u t;
        #pragma unroll
        for (int v = 0; v < 8; v++)
          t[v] = Bs32[s * 16 + 8 * h + v][no];
        fb[j] = __builtin_bit_cast(v16bf, t);
      }
      #pragma unroll
      for (int i = 0; i < 4; i++)
        #pragma unroll
        for (int j = 0; j < 2; j++)
          acc[i][j] = __builtin_amdgcn_wmma_f32_16x16x32_bf16(
              false, fa[i], false, fb[j], (short)0, acc[i][j], false, false);
    }
    __syncthreads();
  }

  // ---- epilogue: lanes0-15 -> M=v, lanes16-31 -> M=8+v; N=lane%16 ----
  for (int i = 0; i < 4; i++) {
    for (int j = 0; j < 2; j++) {
      int gn  = tileN + waveCol * 32 + j * 16 + ml;
      int gm0 = tileM + waveRow * 64 + i * 16 + h * 8;
      if (gn < N) {
        float bv = bias ? bias[gn] : 0.0f;
        #pragma unroll
        for (int v = 0; v < 8; v++) {
          int gm = gm0 + v;
          float r = acc[i][j][v] * alpha + bv;
          if (relu && r < 0.0f) r = 0.0f;
          C[baseC + (long)gm * ldc + gn] = r;
        }
      }
    }
  }
}

// ---------------- elementwise / row kernels ----------------
__global__ void embed_kernel(const int* __restrict__ tokens,
                             const float* __restrict__ tokE,
                             const float* __restrict__ posE,
                             float* __restrict__ x) {
  int row = blockIdx.x;                 // 0..MTOT-1
  int s   = row % SEQ;
  long tb = (long)tokens[row] * EMB;
  long pb = (long)s * EMB;
  long xb = (long)row * EMB;
  for (int e = threadIdx.x; e < EMB; e += blockDim.x)
    x[xb + e] = tokE[tb + e] + posE[pb + e];
}

__global__ void cvt_f32_bf16(const float* __restrict__ in,
                             unsigned short* __restrict__ out, long n) {
  long i = (long)blockIdx.x * blockDim.x + threadIdx.x;
  long st = (long)gridDim.x * blockDim.x;
  for (; i < n; i += st) out[i] = f2bf(in[i]);
}

__global__ void add_kernel(const float* __restrict__ a,
                           const float* __restrict__ b,
                           float* __restrict__ o, long n) {
  long i = (long)blockIdx.x * blockDim.x + threadIdx.x;
  long st = (long)gridDim.x * blockDim.x;
  for (; i < n; i += st) o[i] = a[i] + b[i];
}

// rn = RMSNorm(a (+ b), scale), row length EMB=1024, 256 threads x 4 elems
__global__ void add_rmsnorm_kernel(const float* __restrict__ a,
                                   const float* __restrict__ b,
                                   const float* __restrict__ scale,
                                   float* __restrict__ o) {
  __shared__ float red[256];
  int tid = threadIdx.x;
  long base = (long)blockIdx.x * EMB;
  float v[4]; float ss = 0.0f;
  #pragma unroll
  for (int e = 0; e < 4; e++) {
    long idx = base + tid + e * 256;
    float t = a[idx] + (b ? b[idx] : 0.0f);
    v[e] = t; ss += t * t;
  }
  red[tid] = ss; __syncthreads();
  for (int s2 = 128; s2 > 0; s2 >>= 1) {
    if (tid < s2) red[tid] += red[tid + s2];
    __syncthreads();
  }
  float inv = rsqrtf(red[0] / (float)EMB + 1e-6f);
  #pragma unroll
  for (int e = 0; e < 4; e++) {
    int col = tid + e * 256;
    o[base + col] = v[e] * inv * scale[col];
  }
}

// softmax over rows of length SEQ=1024 (UNMASKED, per reference bug), write bf16
__global__ void softmax_rows_bf16(const float* __restrict__ in,
                                  unsigned short* __restrict__ out) {
  __shared__ float red[256];
  int tid = threadIdx.x;
  long base = (long)blockIdx.x * SEQ;
  float v[4]; float mx = -3.4e38f;
  #pragma unroll
  for (int e = 0; e < 4; e++) {
    v[e] = in[base + tid + e * 256];
    mx = fmaxf(mx, v[e]);
  }
  red[tid] = mx; __syncthreads();
  for (int s2 = 128; s2 > 0; s2 >>= 1) {
    if (tid < s2) red[tid] = fmaxf(red[tid], red[tid + s2]);
    __syncthreads();
  }
  float m = red[0]; __syncthreads();
  float sum = 0.0f;
  #pragma unroll
  for (int e = 0; e < 4; e++) { v[e] = expf(v[e] - m); sum += v[e]; }
  red[tid] = sum; __syncthreads();
  for (int s2 = 128; s2 > 0; s2 >>= 1) {
    if (tid < s2) red[tid] += red[tid + s2];
    __syncthreads();
  }
  float inv = 1.0f / red[0];
  #pragma unroll
  for (int e = 0; e < 4; e++)
    out[base + tid + e * 256] = f2bf(v[e] * inv);
}

// in-place log-softmax over rows of length VOC
__global__ void log_softmax_rows(float* __restrict__ x) {
  __shared__ float red[256];
  int tid = threadIdx.x;
  long base = (long)blockIdx.x * VOC;
  float mx = -3.4e38f;
  for (int c = tid; c < VOC; c += 256) mx = fmaxf(mx, x[base + c]);
  red[tid] = mx; __syncthreads();
  for (int s2 = 128; s2 > 0; s2 >>= 1) {
    if (tid < s2) red[tid] = fmaxf(red[tid], red[tid + s2]);
    __syncthreads();
  }
  float m = red[0]; __syncthreads();
  float sum = 0.0f;
  for (int c = tid; c < VOC; c += 256) sum += expf(x[base + c] - m);
  red[tid] = sum; __syncthreads();
  for (int s2 = 128; s2 > 0; s2 >>= 1) {
    if (tid < s2) red[tid] += red[tid + s2];
    __syncthreads();
  }
  float lse = m + logf(red[0]);
  for (int c = tid; c < VOC; c += 256) x[base + c] -= lse;
}

// ---------------- host orchestration ----------------
extern "C" void kernel_launch(void* const* d_in, const int* in_sizes, int n_in,
                              void* d_out, int out_size, void* d_ws, size_t ws_size,
                              hipStream_t stream) {
  (void)in_sizes; (void)n_in; (void)out_size; (void)ws_size;

  const int*   tokens = (const int*)  d_in[0];
  const float* tokE   = (const float*)d_in[1];
  const float* posE   = (const float*)d_in[2];
  const float* Qw     = (const float*)d_in[3];
  const float* Kw     = (const float*)d_in[4];
  const float* Vw     = (const float*)d_in[5];
  const float* Ow     = (const float*)d_in[6];
  const float* rs2    = (const float*)d_in[7];
  const float* w1     = (const float*)d_in[8];
  const float* b1     = (const float*)d_in[9];
  const float* w2     = (const float*)d_in[10];
  const float* b2     = (const float*)d_in[11];
  const float* frs    = (const float*)d_in[12];
  const float* w_out  = (const float*)d_in[13];
  const float* b_out  = (const float*)d_in[14];
  float* out = (float*)d_out;

  // deterministic bump allocator over d_ws (~450 MB total)
  char* ws = (char*)d_ws;
  size_t off = 0;
  auto alloc = [&](size_t bytes) -> char* {
    char* p = ws + off;
    off += (bytes + 255) & ~(size_t)255;
    return p;
  };

  float* x    = (float*)alloc((size_t)MTOT * EMB * 4);          // activations
  float* h1   = (float*)alloc((size_t)MTOT * FFD * 4);          // FFN hidden
  float* sc   = (float*)alloc((size_t)BAT * NH * SEQ * SEQ * 4);// scores fp32
  unsigned short* attnb = (unsigned short*)alloc((size_t)BAT * NH * SEQ * SEQ * 2);
  float* bufA = (float*)alloc((size_t)MTOT * HID * 4);          // q / mha
  float* bufB = (float*)alloc((size_t)MTOT * HID * 4);          // k / proj
  float* bufC = (float*)alloc((size_t)MTOT * EMB * 4);          // v / rn / ff
  unsigned short* actb = (unsigned short*)alloc((size_t)MTOT * EMB * 2); // xb/mhab/rnb
  unsigned short* qb   = (unsigned short*)alloc((size_t)MTOT * HID * 2);
  unsigned short* kb   = (unsigned short*)alloc((size_t)MTOT * HID * 2);
  unsigned short* vb   = (unsigned short*)alloc((size_t)MTOT * HID * 2);
  unsigned short* h1b  = (unsigned short*)alloc((size_t)MTOT * FFD * 2);
  unsigned short* wQb  = (unsigned short*)alloc((size_t)LNUM * EMB * HID * 2);
  unsigned short* wKb  = (unsigned short*)alloc((size_t)LNUM * EMB * HID * 2);
  unsigned short* wVb  = (unsigned short*)alloc((size_t)LNUM * EMB * HID * 2);
  unsigned short* wOb  = (unsigned short*)alloc((size_t)LNUM * HID * EMB * 2);
  unsigned short* w1b  = (unsigned short*)alloc((size_t)LNUM * EMB * FFD * 2);
  unsigned short* w2b  = (unsigned short*)alloc((size_t)LNUM * FFD * EMB * 2);
  unsigned short* wob  = (unsigned short*)alloc((size_t)EMB * VOC * 2);

  auto cvt = [&](const float* in, unsigned short* o, long n) {
    cvt_f32_bf16<<<4096, 256, 0, stream>>>(in, o, n);
  };
  auto gemm = [&](const unsigned short* A, const unsigned short* Bm, float* C,
                  const float* bias, int M, int N, int K,
                  long baoA, long baiA, long rsA, long csA,
                  long baoB, long baiB, long rsB, long csB,
                  long baoC, long baiC, long ldc,
                  int nInner, int nBatch, float alpha, int relu) {
    dim3 g((N + 127) / 128, (M + 127) / 128, nBatch);
    gemm_bf16_wmma<<<g, 256, 0, stream>>>(A, Bm, C, bias, M, N, K,
                                          baoA, baiA, rsA, csA,
                                          baoB, baiB, rsB, csB,
                                          baoC, baiC, ldc, nInner, alpha, relu);
  };

  // weights -> bf16 once per call (bf16 copies then live mostly in 192MB L2)
  cvt(Qw, wQb, (long)LNUM * EMB * HID);
  cvt(Kw, wKb, (long)LNUM * EMB * HID);
  cvt(Vw, wVb, (long)LNUM * EMB * HID);
  cvt(Ow, wOb, (long)LNUM * HID * EMB);
  cvt(w1, w1b, (long)LNUM * EMB * FFD);
  cvt(w2, w2b, (long)LNUM * FFD * EMB);
  cvt(w_out, wob, (long)EMB * VOC);

  // x = tok_embed[tokens] + pos_embed
  embed_kernel<<<MTOT, 256, 0, stream>>>(tokens, tokE, posE, x);

  const float iscale = 1.0f / 8.0f;  // 1/sqrt(HD=64)
  for (int l = 0; l < LNUM; l++) {
    const unsigned short* Qb = wQb + (size_t)l * EMB * HID;
    const unsigned short* Kb = wKb + (size_t)l * EMB * HID;
    const unsigned short* Vb = wVb + (size_t)l * EMB * HID;
    const unsigned short* Ob = wOb + (size_t)l * HID * EMB;
    const unsigned short* W1 = w1b + (size_t)l * EMB * FFD;
    const unsigned short* W2 = w2b + (size_t)l * FFD * EMB;

    // NOTE (bug-faithful): QKV act on raw x, not rmsnorm(x)
    cvt(x, actb, (long)MTOT * EMB);
    gemm(actb, Qb, bufA, nullptr, MTOT, HID, EMB, 0,0,EMB,1, 0,0,HID,1, 0,0,HID, 1,1, 1.0f, 0);
    gemm(actb, Kb, bufB, nullptr, MTOT, HID, EMB, 0,0,EMB,1, 0,0,HID,1, 0,0,HID, 1,1, 1.0f, 0);
    gemm(actb, Vb, bufC, nullptr, MTOT, HID, EMB, 0,0,EMB,1, 0,0,HID,1, 0,0,HID, 1,1, 1.0f, 0);
    cvt(bufA, qb, (long)MTOT * HID);
    cvt(bufB, kb, (long)MTOT * HID);
    cvt(bufC, vb, (long)MTOT * HID);

    // scores[b,h,s,t] = q.k^T * scale  (batched over z = b*NH + h)
    gemm(qb, kb, sc, nullptr, SEQ, SEQ, HD,
         (long)SEQ * HID, HD, HID, 1,
         (long)SEQ * HID, HD, 1, HID,
         (long)NH * SEQ * SEQ, (long)SEQ * SEQ, SEQ,
         NH, BAT * NH, iscale, 0);

    // softmax (UNMASKED, per reference bug) -> bf16 attn
    softmax_rows_bf16<<<BAT * NH * SEQ, 256, 0, stream>>>(sc, attnb);

    // mha[b,t,h,d] = sum_s attn[b,h,s,t] * v[b,s,h,d]   (A = attn^T)
    gemm(attnb, vb, bufA, nullptr, SEQ, HD, SEQ,
         (long)NH * SEQ * SEQ, (long)SEQ * SEQ, 1, SEQ,
         (long)SEQ * HID, HD, HID, 1,
         (long)SEQ * HID, HD, HID,
         NH, BAT * NH, 1.0f, 0);

    // proj = mha @ O
    cvt(bufA, actb, (long)MTOT * HID);
    gemm(actb, Ob, bufB, nullptr, MTOT, EMB, HID, 0,0,HID,1, 0,0,EMB,1, 0,0,EMB, 1,1, 1.0f, 0);

    // rn = RMSNorm(x + proj)
    add_rmsnorm_kernel<<<MTOT, 256, 0, stream>>>(x, bufB, rs2 + (size_t)l * EMB, bufC);

    // ff = relu(rn@w1+b1) @ w2 + b2
    cvt(bufC, actb, (long)MTOT * EMB);
    gemm(actb, W1, h1, b1 + (size_t)l * FFD, MTOT, FFD, EMB, 0,0,EMB,1, 0,0,FFD,1, 0,0,FFD, 1,1, 1.0f, 1);
    cvt(h1, h1b, (long)MTOT * FFD);
    gemm(h1b, W2, bufC, b2 + (size_t)l * EMB, MTOT, EMB, FFD, 0,0,FFD,1, 0,0,EMB,1, 0,0,EMB, 1,1, 1.0f, 0);

    // x = proj + ff (bug-faithful: input x is NOT re-added)
    add_kernel<<<4096, 256, 0, stream>>>(bufB, bufC, x, (long)MTOT * EMB);
  }

  // final rmsnorm, logits, log_softmax (in-place on d_out)
  add_rmsnorm_kernel<<<MTOT, 256, 0, stream>>>(x, nullptr, frs, bufC);
  cvt(bufC, actb, (long)MTOT * EMB);
  gemm(actb, wob, out, b_out, MTOT, VOC, EMB, 0,0,EMB,1, 0,0,VOC,1, 0,0,VOC, 1,1, 1.0f, 0);
  log_softmax_rows<<<MTOT, 256, 0, stream>>>(out);
}
